// CAMD_48936857370892
// MI455X (gfx1250) — matrix operand order repaired and checked by reading, not compiled, not run on hardware
//
#include <hip/hip_runtime.h>
#include <hip/hip_bf16.h>

// Problem constants (match reference)
constexpr int MM   = 4;      // modalities
constexpr int TT   = 8192;   // time steps
constexpr int DD   = 64;     // feature dim
constexpr int NC   = 64;     // scan chunks
constexpr int CL   = TT/NC;  // chunk length = 128

typedef __attribute__((ext_vector_type(16))) _Float16 v16h;
typedef __attribute__((ext_vector_type(8)))  _Float16 v8h;
typedef __attribute__((ext_vector_type(8)))  float    v8f;

// ---------------------------------------------------------------------------
// Kernel 1: fused 3-layer MLP (x -> relu(xW0+b0) -> relu(xW1+b1) -> xW2+b2)
// One block = 8 waves, each wave owns a 16-row tile; blockIdx.y selects head:
//   head 0      -> Q  = mlp(X[0], wq_w, wq_b)
//   head 1..4   -> K_m = mlp(X[m], wk_w[m], wk_b[m])
// Input tile staged with CDNA5 async global->LDS copies (ASYNCcnt), GEMMs use
// v_wmma_f32_16x16x32_f16 (wave32 WMMA), f32 accumulate.
// ---------------------------------------------------------------------------
__global__ void __launch_bounds__(256)
camd_mlp_wmma(const float* __restrict__ X,
              const float* __restrict__ wq_w, const float* __restrict__ wq_b,
              const float* __restrict__ wk_w, const float* __restrict__ wk_b,
              float* __restrict__ qbuf, float* __restrict__ kbuf)
{
    // Weights transposed in LDS: wT[layer][n][k] = W[layer][k][n]  (f16)
    __shared__ __align__(32) _Float16 wT[3][64][64];   // 24 KB
    __shared__ float                  biasS[3][64];    // 768 B
    __shared__ __align__(32) float    actF[8][16][64]; // 32 KB (f32 tiles)

    const int tid  = threadIdx.x;
    const int head = blockIdx.y;
    const int mIn  = (head == 0) ? 0 : (head - 1);
    const float* Wg = (head == 0) ? wq_w : (wk_w + (size_t)(head - 1) * 3 * 64 * 64);
    const float* Bg = (head == 0) ? wq_b : (wk_b + (size_t)(head - 1) * 3 * 64);
    float* outBuf   = (head == 0) ? qbuf : (kbuf + (size_t)(head - 1) * TT * 64);
    const float* inPtr = X + (size_t)mIn * TT * 64;
    const int blockRow0 = blockIdx.x * 128;

    // --- Async copy: 128 rows x 64 f32 (32 KB) global -> LDS, 16B per lane/op.
    // LDS byte address = low 32 bits of the generic pointer (ISA 10.2: flat
    // LDS addresses truncate to addr[31:0]).
    {
        const unsigned actBase = (unsigned)(size_t)&actF[0][0][0];
        const float* gBase = inPtr + (size_t)blockRow0 * 64;
#pragma unroll
        for (int k2 = 0; k2 < 8; ++k2) {
            const unsigned fofs = (unsigned)tid * 4u + (unsigned)k2 * 1024u;
            const unsigned lds  = actBase + fofs * 4u;
            const unsigned long long g =
                (unsigned long long)(size_t)(gBase + fofs);
            asm volatile("global_load_async_to_lds_b128 %0, %1, off"
                         :: "v"(lds), "v"(g) : "memory");
        }
    }

    // Cooperative load: weights (transposed, f32->f16) and biases (VALU path:
    // transform in flight, so not a raw byte copy)
    for (int idx = tid; idx < 3 * 64 * 64; idx += 256) {
        int l = idx >> 12, rem = idx & 4095, k = rem >> 6, n = rem & 63;
        wT[l][n][k] = (_Float16)Wg[idx];
    }
    for (int idx = tid; idx < 3 * 64; idx += 256)
        biasS[idx >> 6][idx & 63] = Bg[idx];

    asm volatile("s_wait_asynccnt 0x0" ::: "memory");
    __syncthreads();

    const int wv     = tid >> 5;        // wave id 0..7
    const int lane   = tid & 31;
    const int hiHalf = (lane >= 16) ? 1 : 0;
    const int col    = lane & 15;       // N within 16-wide tile / C-matrix col
    const int arow   = lane & 15;       // A-matrix row for this lane
    const int rowBase = blockRow0 + wv * 16;

    for (int layer = 0; layer < 3; ++layer) {
        // --- A fragments (16x32 f16, two K-chunks), contiguous layout:
        // lane<16: K = j ; lane>=16: K = j+16   (chunk base +0/+32)
        v16h a0, a1;
        {
            const int base = hiHalf * 16;
            v8f lo0 = *(const v8f*)&actF[wv][arow][base];
            v8f hi0 = *(const v8f*)&actF[wv][arow][base + 8];
            v8f lo1 = *(const v8f*)&actF[wv][arow][base + 32];
            v8f hi1 = *(const v8f*)&actF[wv][arow][base + 40];
#pragma unroll
            for (int j = 0; j < 8; ++j) {
                a0[j] = (_Float16)lo0[j]; a0[j + 8] = (_Float16)hi0[j];
                a1[j] = (_Float16)lo1[j]; a1[j + 8] = (_Float16)hi1[j];
            }
        }

        // --- Preload all B fragments (32x16 f16): lane<16 -> K=j, else K=j+16;
        //     element j reads wT[layer][N][k0 + hiHalf*16 + j] (contiguous 32B)
        v16h bfr[4][2];
#pragma unroll
        for (int n = 0; n < 4; ++n) {
            const int N = n * 16 + col;
            bfr[n][0] = *(const v16h*)&wT[layer][N][hiHalf * 16];
            bfr[n][1] = *(const v16h*)&wT[layer][N][32 + hiHalf * 16];
        }

        v8f acc[4];
#pragma unroll
        for (int n = 0; n < 4; ++n) acc[n] = (v8f){};

        // --- dense WMMA block: 8 back-to-back v_wmma_f32_16x16x32_f16
#pragma unroll
        for (int n = 0; n < 4; ++n) {
            acc[n] = __builtin_amdgcn_wmma_f32_16x16x32_f16(
                false, a0, false, bfr[n][0], (short)0, acc[n], false, false);
            acc[n] = __builtin_amdgcn_wmma_f32_16x16x32_f16(
                false, a1, false, bfr[n][1], (short)0, acc[n], false, false);
        }

        // --- epilogue: C/D layout VGPR r -> row = r + hiHalf*8, col = lane&15
#pragma unroll
        for (int n = 0; n < 4; ++n) {
#pragma unroll
            for (int r = 0; r < 8; ++r) {
                const int mr = r + hiHalf * 8;
                const int nc = n * 16 + col;
                float v = acc[n][r] + biasS[layer][nc];
                if (layer < 2) {
                    v = v > 0.0f ? v : 0.0f;
                    actF[wv][mr][nc] = v;
                } else {
                    outBuf[(size_t)(rowBase + mr) * 64 + nc] = v;
                }
            }
        }
        __syncthreads();
    }
}

// ---------------------------------------------------------------------------
// Kernel 2: per-chunk partial sums  part[m][c][e*64+d] = sum_chunk K[t,d]*V[t,e]
// ---------------------------------------------------------------------------
__global__ void __launch_bounds__(128)
camd_chunk_partial(const float* __restrict__ X, const float* __restrict__ kbuf,
                   float* __restrict__ part)
{
    const int c = blockIdx.x, m = blockIdx.y;
    const int t = threadIdx.x, d = t & 63, e = t >> 6;
    const float* Kp = kbuf + ((size_t)m * TT + (size_t)c * CL) * 64;
    const float* Vp = X    + ((size_t)m * TT + (size_t)c * CL) * 64;
    float acc = 0.0f;
    for (int j = 0; j < CL; ++j) {
        __builtin_prefetch(Kp + (j + 8) * 64 + d, 0, 0);  // global_prefetch_b8
        acc += Kp[j * 64 + d] * Vp[j * 64 + e];
    }
    part[(((size_t)m * NC) + c) * 128 + t] = acc;
}

// ---------------------------------------------------------------------------
// Kernel 3: serial exclusive scan over the NC chunk sums (per m,d,e)
// ---------------------------------------------------------------------------
__global__ void __launch_bounds__(128)
camd_chunk_scan(float* __restrict__ part)
{
    const int m = blockIdx.x, t = threadIdx.x;
    float run = 0.0f;
    for (int c = 0; c < NC; ++c) {
        const size_t idx = (((size_t)m * NC) + c) * 128 + t;
        const float v = part[idx];
        part[idx] = run;
        run += v;
    }
}

// ---------------------------------------------------------------------------
// Kernel 4: materialize inclusive state S[m][t][e*64+d] with chunk prefixes
// ---------------------------------------------------------------------------
__global__ void __launch_bounds__(128)
camd_state(const float* __restrict__ X, const float* __restrict__ kbuf,
           const float* __restrict__ part, float* __restrict__ S)
{
    const int c = blockIdx.x, m = blockIdx.y;
    const int t = threadIdx.x, d = t & 63, e = t >> 6;
    const float* Kp = kbuf + ((size_t)m * TT + (size_t)c * CL) * 64;
    const float* Vp = X    + ((size_t)m * TT + (size_t)c * CL) * 64;
    float acc = part[(((size_t)m * NC) + c) * 128 + t];
    float* Sp = S + ((size_t)m * TT + (size_t)c * CL) * 128;
    for (int j = 0; j < CL; ++j) {
        acc += Kp[j * 64 + d] * Vp[j * 64 + e];
        Sp[j * 128 + t] = acc;
    }
}

// ---------------------------------------------------------------------------
// Kernel 5: per-query gather + contraction. One wave per query:
//   idx = searchsorted(t2m, t1[i], 'right') - 1 ;  out[i,e] = sum_d Q[i,d]*S[idx,d,e]
// ---------------------------------------------------------------------------
__global__ void __launch_bounds__(256)
camd_query(const float* __restrict__ X, const float* __restrict__ qbuf,
           const float* __restrict__ S, float* __restrict__ out)
{
    const int wv = threadIdx.x >> 5, lane = threadIdx.x & 31;
    const int i = blockIdx.x * 8 + wv;
    const float t1 = X[(size_t)i * 64 + 63];
    const float q0 = qbuf[(size_t)i * 64 + lane];
    const float q1 = qbuf[(size_t)i * 64 + lane + 32];
    float p0 = 0.0f, p1 = 0.0f;
    for (int m = 0; m < MM; ++m) {
        const float* t2 = X + (size_t)m * TT * 64 + 63;  // stride 64 floats
        int lo = 0, hi = TT;
        while (lo < hi) {                 // uniform per wave: no divergence
            const int mid = (lo + hi) >> 1;
            if (t2[(size_t)mid * 64] <= t1) lo = mid + 1; else hi = mid;
        }
        const int idx = lo - 1;
        if (idx >= 0) {
            const float* Sp = S + ((size_t)m * TT + idx) * 128;
            p0 += q0 * Sp[lane]      + q1 * Sp[lane + 32];
            p1 += q0 * Sp[64 + lane] + q1 * Sp[96 + lane];
        }
    }
#pragma unroll
    for (int off = 16; off > 0; off >>= 1) {
        p0 += __shfl_down(p0, off, 32);
        p1 += __shfl_down(p1, off, 32);
    }
    if (lane == 0) {
        out[(size_t)i * 2 + 0] = p0;
        out[(size_t)i * 2 + 1] = p1;
    }
}

// ---------------------------------------------------------------------------
extern "C" void kernel_launch(void* const* d_in, const int* in_sizes, int n_in,
                              void* d_out, int out_size, void* d_ws, size_t ws_size,
                              hipStream_t stream)
{
    const float* X    = (const float*)d_in[0];  // (M,T,D)
    const float* wq_w = (const float*)d_in[1];  // (3,D,D)
    const float* wq_b = (const float*)d_in[2];  // (3,D)
    const float* wk_w = (const float*)d_in[3];  // (M,3,D,D)
    const float* wk_b = (const float*)d_in[4];  // (M,3,D)
    float* out = (float*)d_out;                 // (1,T,2)
    float* ws  = (float*)d_ws;

    // Workspace layout (floats): Q | K | chunk partials | S   (~27.4 MB total)
    float* qbuf = ws;                                   // T*64
    float* kbuf = qbuf + (size_t)TT * 64;               // M*T*64
    float* part = kbuf + (size_t)MM * TT * 64;          // M*NC*128
    float* Sbuf = part + (size_t)MM * NC * 128;         // M*T*128

    camd_mlp_wmma   <<<dim3(TT / 128, 5), 256, 0, stream>>>(X, wq_w, wq_b, wk_w, wk_b, qbuf, kbuf);
    camd_chunk_partial<<<dim3(NC, MM),    128, 0, stream>>>(X, kbuf, part);
    camd_chunk_scan <<<MM,                128, 0, stream>>>(part);
    camd_state      <<<dim3(NC, MM),      128, 0, stream>>>(X, kbuf, part, Sbuf);
    camd_query      <<<TT / 8,            256, 0, stream>>>(X, qbuf, Sbuf, out);
}
